// MultiheadAttention_76373108458020
// MI455X (gfx1250) — matrix operand order repaired and checked by reading, not compiled, not run on hardware
//
#include <hip/hip_runtime.h>

// ---------------------------------------------------------------------------
// MultiheadAttention forward for MI455X (gfx1250, wave32, WMMA)
// B=4, S=2048, D=768, H=12, HD=64. bf16 WMMA (16x16x32) with f32 accum.
// Double-buffered async global->LDS staging (ASYNCcnt) overlapped with WMMA.
// ---------------------------------------------------------------------------

typedef __attribute__((ext_vector_type(16))) __bf16 v16bf;
typedef __attribute__((ext_vector_type(8)))  float  v8f;

constexpr int NB  = 4;
constexpr int NS  = 2048;
constexpr int ND  = 768;
constexpr int NE  = 3 * ND;   // 2304
constexpr int NH  = 12;
constexpr int NHD = 64;
constexpr int NBS = NB * NS;  // 8192 token rows

// ---- async global -> LDS 16-byte copy (per lane), tracked by ASYNCcnt ------
__device__ __forceinline__ void async_ld16(const __bf16* lds, const __bf16* g) {
  unsigned int       l  = (unsigned int)(uintptr_t)lds;        // LDS offset
  unsigned long long ga = (unsigned long long)(uintptr_t)g;    // global addr
  asm volatile("global_load_async_to_lds_b128 %0, %1, off"
               :: "v"(l), "v"(ga) : "memory");
}
template <int N>
__device__ __forceinline__ void wait_async_le() {
  asm volatile("s_wait_asynccnt %0" :: "n"(N) : "memory");
}

// ---- WMMA fragment helpers (layouts per CDNA5 ISA 7.12.2) ------------------
// A 16x32 bf16: lane (hq = lane/16, row = lane%16) holds
//   K = hq*8 + [0..7]  (vgpr0-3)  and  K = 16 + hq*8 + [0..7] (vgpr4-7)
__device__ __forceinline__ v16bf frag_a(const __bf16* rowbase, int hq) {
  union { uint4 u[2]; v16bf v; } f;
  f.u[0] = *(const uint4*)(rowbase + hq * 8);
  f.u[1] = *(const uint4*)(rowbase + 16 + hq * 8);
  return f.v;
}
// B 32x16 bf16: lane (hq, col = lane%16) holds column `col`,
//   K = hq*8 + [0..7]  and  K = 16 + hq*8 + [0..7]  (same order as A)
__device__ __forceinline__ v16bf frag_b(const __bf16* colbase, int hq) {
  union { uint4 u[2]; v16bf v; } f;
  f.u[0] = *(const uint4*)(colbase + hq * 8);
  f.u[1] = *(const uint4*)(colbase + 16 + hq * 8);
  return f.v;
}
__device__ __forceinline__ v8f wmma_bf16(v16bf a, v16bf b, v8f c) {
  return __builtin_amdgcn_wmma_f32_16x16x32_bf16(false, a, false, b,
                                                 (short)0, c, false, false);
}

// ---- Kernel 0: fp32 -> bf16 conversion -------------------------------------
__global__ __launch_bounds__(256) void cvt_f32_bf16(
    const float* __restrict__ src, __bf16* __restrict__ dst, int n) {
  int i = blockIdx.x * 256 + threadIdx.x;
  if (i < n) dst[i] = (__bf16)src[i];
}

// ---- Kernel 1: QKV projection GEMM + head split ----------------------------
// qkv[row, e] = sum_d x[row,d] * w_qkv[e,d] + b_qkv[e]
// e = h*192 + which*64 + d ; q scaled by 0.125 ; out layout [B,H,S,64]
__global__ __launch_bounds__(256) void qkv_gemm(
    const __bf16* __restrict__ xb, const __bf16* __restrict__ wb,
    const float* __restrict__ bias,
    __bf16* __restrict__ qo, __bf16* __restrict__ ko, __bf16* __restrict__ vo) {
  __shared__ __align__(16) __bf16 At[2][128][32];
  __shared__ __align__(16) __bf16 Bt[2][128][32];

  const int t    = threadIdx.x;
  const int wave = t >> 5;
  const int lane = t & 31;
  const int hq   = lane >> 4;
  const int col  = lane & 15;
  const int row0 = blockIdx.x * 128;
  const int n0   = blockIdx.y * 128;

  v8f acc[8];
#pragma unroll
  for (int i = 0; i < 8; ++i) acc[i] = {};

  const int sr = t >> 1;          // staging row 0..127
  const int sh = (t & 1) * 16;    // staging k-offset 0 or 16
  const __bf16* ag = xb + (size_t)(row0 + sr) * ND + sh;
  const __bf16* bg = wb + (size_t)(n0 + sr) * ND + sh;

  auto stage = [&](int k0, int buf) {
    async_ld16(&At[buf][sr][sh],     ag + k0);
    async_ld16(&At[buf][sr][sh + 8], ag + k0 + 8);
    async_ld16(&Bt[buf][sr][sh],     bg + k0);
    async_ld16(&Bt[buf][sr][sh + 8], bg + k0 + 8);
  };

  constexpr int NIT = ND / 32;    // 24 k-tiles
  stage(0, 0);
  for (int it = 0; it < NIT; ++it) {
    const int cur = it & 1;
    if (it + 1 < NIT) {
      stage((it + 1) * 32, cur ^ 1);           // copy for next tile in flight
      __builtin_prefetch(ag + (it + 2) * 32, 0, 1);
      __builtin_prefetch(bg + (it + 2) * 32, 0, 1);
      wait_async_le<4>();                      // oldest tile (cur) has landed
    } else {
      wait_async_le<0>();
    }
    __syncthreads();
    v16bf af = frag_a(&At[cur][wave * 16 + col][0], hq);
#pragma unroll
    for (int nt = 0; nt < 8; ++nt) {
      v16bf bf = frag_b(&Bt[cur][nt * 16 + col][0], hq);
      acc[nt] = wmma_bf16(af, bf, acc[nt]);
    }
    __syncthreads();
  }

  const int m_lo = hq * 8;
#pragma unroll
  for (int nt = 0; nt < 8; ++nt) {
    const int e     = n0 + nt * 16 + col;
    const int h     = e / 192;
    const int inner = e % 192;
    const int which = inner >> 6;
    const int d     = inner & 63;
    const float bv  = bias[e];
    __bf16* dstb    = (which == 0) ? qo : (which == 1) ? ko : vo;
    const float scl = (which == 0) ? 0.125f : 1.0f;  // fold 1/sqrt(64) into q
#pragma unroll
    for (int r = 0; r < 8; ++r) {
      const int row = row0 + wave * 16 + m_lo + r;
      const int b   = row >> 11;     // / 2048
      const int s   = row & 2047;
      float v = (acc[nt][r] + bv) * scl;
      dstb[((size_t)(b * NH + h) * NS + s) * NHD + d] = (__bf16)v;
    }
  }
}

// ---- Kernel 2: flash attention per (b,h), 128 query rows / block -----------
__global__ __launch_bounds__(256) void attn_kernel(
    const __bf16* __restrict__ qg, const __bf16* __restrict__ kg,
    const __bf16* __restrict__ vg, __bf16* __restrict__ og) {
  __shared__ __align__(16) __bf16 Kt[2][64][64];   // keys x head-dim
  __shared__ __align__(16) __bf16 Vt[2][64][64];   // head-dim x keys (transposed)
  __shared__ __align__(16) __bf16 Pl[8][16][64];   // per-wave P staging

  const int t    = threadIdx.x;
  const int wave = t >> 5;
  const int lane = t & 31;
  const int hq   = lane >> 4;
  const int col  = lane & 15;
  const int q0   = blockIdx.x * 128;
  const int bh   = blockIdx.y;            // b*NH + h
  const int b    = bh / NH;
  const int h    = bh % NH;

  const size_t head_off = (size_t)bh * NS * NHD;
  const __bf16* qp = qg + head_off;
  const __bf16* kp = kg + head_off;
  const __bf16* vp = vg + head_off;

  // Q fragments (K-dim 64 -> two 16x32 A fragments), resident in registers
  const int qrow = q0 + wave * 16 + col;
  const v16bf qf0 = frag_a(qp + (size_t)qrow * NHD, hq);
  const v16bf qf1 = frag_a(qp + (size_t)qrow * NHD + 32, hq);

  v8f oacc[4];
#pragma unroll
  for (int i = 0; i < 4; ++i) oacc[i] = {};
  float mstat[8], lstat[8];
#pragma unroll
  for (int r = 0; r < 8; ++r) { mstat[r] = -INFINITY; lstat[r] = 0.0f; }

  const int skey = t >> 2;          // staging key row 0..63
  const int sq4  = (t & 3) * 16;    // staging dim offset 0/16/32/48

  auto stageK = [&](int kb0, int buf) {
    async_ld16(&Kt[buf][skey][sq4],     kp + (size_t)(kb0 + skey) * NHD + sq4);
    async_ld16(&Kt[buf][skey][sq4 + 8], kp + (size_t)(kb0 + skey) * NHD + sq4 + 8);
  };
  auto stageV = [&](int kb0, int buf) {
#pragma unroll
    for (int i = 0; i < 2; ++i) {
      const int idx = t + i * 256;       // 0..511
      const int key = idx & 63;
      const int dg  = (idx >> 6) * 8;    // 0,8,...,56
      union { uint4 u; __bf16 e[8]; } vv;
      vv.u = *(const uint4*)(vp + (size_t)(kb0 + key) * NHD + dg);
#pragma unroll
      for (int j = 0; j < 8; ++j) Vt[buf][dg + j][key] = vv.e[j];
    }
  };

  constexpr int NIT = NS / 64;      // 32 key tiles
  stageK(0, 0);
  stageV(0, 0);
  for (int it = 0; it < NIT; ++it) {
    const int cur = it & 1;
    if (it + 1 < NIT) {
      stageK((it + 1) * 64, cur ^ 1);   // next K tile in flight during compute
      stageV((it + 1) * 64, cur ^ 1);
      wait_async_le<2>();               // current K tile has landed
    } else {
      wait_async_le<0>();
    }
    __syncthreads();

    // scores: S = Qs @ K^T  (Q pre-scaled by 1/8) -------------------------
    v8f st[4];
#pragma unroll
    for (int nt = 0; nt < 4; ++nt) {
      v8f s = {};
      s = wmma_bf16(qf0, frag_b(&Kt[cur][nt * 16 + col][0], hq), s);
      s = wmma_bf16(qf1, frag_b(&Kt[cur][nt * 16 + col][32], hq), s);
      st[nt] = s;
    }

    // online softmax: row stats live across 16-lane halves ----------------
    float mx[8];
#pragma unroll
    for (int r = 0; r < 8; ++r)
      mx[r] = fmaxf(fmaxf(st[0][r], st[1][r]), fmaxf(st[2][r], st[3][r]));
#pragma unroll
    for (int msk = 1; msk < 16; msk <<= 1)
#pragma unroll
      for (int r = 0; r < 8; ++r) mx[r] = fmaxf(mx[r], __shfl_xor(mx[r], msk));

    float alpha[8], rs[8];
#pragma unroll
    for (int r = 0; r < 8; ++r) {
      const float mn = fmaxf(mstat[r], mx[r]);
      alpha[r] = __expf(mstat[r] - mn);
      mstat[r] = mn;
      rs[r] = 0.0f;
    }
#pragma unroll
    for (int nt = 0; nt < 4; ++nt)
#pragma unroll
      for (int r = 0; r < 8; ++r) {
        const float p = __expf(st[nt][r] - mstat[r]);
        rs[r] += p;
        Pl[wave][hq * 8 + r][nt * 16 + col] = (__bf16)p;  // C-layout -> LDS
      }
#pragma unroll
    for (int msk = 1; msk < 16; msk <<= 1)
#pragma unroll
      for (int r = 0; r < 8; ++r) rs[r] += __shfl_xor(rs[r], msk);
#pragma unroll
    for (int r = 0; r < 8; ++r) lstat[r] = lstat[r] * alpha[r] + rs[r];
#pragma unroll
    for (int nt = 0; nt < 4; ++nt)
#pragma unroll
      for (int r = 0; r < 8; ++r) oacc[nt][r] *= alpha[r];

    // O += P @ V  (A frags re-read from LDS in A-layout) ------------------
#pragma unroll
    for (int kb = 0; kb < 2; ++kb) {
      const v16bf pa = frag_a(&Pl[wave][col][kb * 32], hq);
#pragma unroll
      for (int nt = 0; nt < 4; ++nt)
        oacc[nt] = wmma_bf16(pa, frag_b(&Vt[cur][nt * 16 + col][kb * 32], hq),
                             oacc[nt]);
    }
    __syncthreads();
  }

  // epilogue: divide by row sums, write merged-head layout [B,S,D] ---------
#pragma unroll
  for (int r = 0; r < 8; ++r) {
    const float inv = 1.0f / lstat[r];
    const int m = hq * 8 + r;
    const int row = b * NS + q0 + wave * 16 + m;
#pragma unroll
    for (int nt = 0; nt < 4; ++nt)
      og[(size_t)row * ND + h * NHD + nt * 16 + col] = (__bf16)(oacc[nt][r] * inv);
  }
}

// ---- Kernel 3: output projection GEMM (fp32 out) ---------------------------
__global__ __launch_bounds__(256) void out_gemm(
    const __bf16* __restrict__ ab, const __bf16* __restrict__ wb,
    const float* __restrict__ bias, float* __restrict__ out) {
  __shared__ __align__(16) __bf16 At[2][128][32];
  __shared__ __align__(16) __bf16 Bt[2][128][32];

  const int t    = threadIdx.x;
  const int wave = t >> 5;
  const int lane = t & 31;
  const int hq   = lane >> 4;
  const int col  = lane & 15;
  const int row0 = blockIdx.x * 128;
  const int n0   = blockIdx.y * 128;

  v8f acc[8];
#pragma unroll
  for (int i = 0; i < 8; ++i) acc[i] = {};

  const int sr = t >> 1;
  const int sh = (t & 1) * 16;
  const __bf16* ag = ab + (size_t)(row0 + sr) * ND + sh;
  const __bf16* bg = wb + (size_t)(n0 + sr) * ND + sh;

  auto stage = [&](int k0, int buf) {
    async_ld16(&At[buf][sr][sh],     ag + k0);
    async_ld16(&At[buf][sr][sh + 8], ag + k0 + 8);
    async_ld16(&Bt[buf][sr][sh],     bg + k0);
    async_ld16(&Bt[buf][sr][sh + 8], bg + k0 + 8);
  };

  constexpr int NIT = ND / 32;    // 24 k-tiles
  stage(0, 0);
  for (int it = 0; it < NIT; ++it) {
    const int cur = it & 1;
    if (it + 1 < NIT) {
      stage((it + 1) * 32, cur ^ 1);
      __builtin_prefetch(ag + (it + 2) * 32, 0, 1);
      __builtin_prefetch(bg + (it + 2) * 32, 0, 1);
      wait_async_le<4>();
    } else {
      wait_async_le<0>();
    }
    __syncthreads();
    v16bf af = frag_a(&At[cur][wave * 16 + col][0], hq);
#pragma unroll
    for (int nt = 0; nt < 8; ++nt) {
      v16bf bf = frag_b(&Bt[cur][nt * 16 + col][0], hq);
      acc[nt] = wmma_bf16(af, bf, acc[nt]);
    }
    __syncthreads();
  }

  const int m_lo = hq * 8;
#pragma unroll
  for (int nt = 0; nt < 8; ++nt) {
    const int e    = n0 + nt * 16 + col;
    const float bv = bias[e];
#pragma unroll
    for (int r = 0; r < 8; ++r) {
      const int row = row0 + wave * 16 + m_lo + r;
      out[(size_t)row * ND + e] = acc[nt][r] + bv;
    }
  }
}

// ---------------------------------------------------------------------------
extern "C" void kernel_launch(void* const* d_in, const int* in_sizes, int n_in,
                              void* d_out, int out_size, void* d_ws, size_t ws_size,
                              hipStream_t stream) {
  (void)in_sizes; (void)n_in; (void)out_size; (void)ws_size;
  const float* x     = (const float*)d_in[0];
  const float* w_qkv = (const float*)d_in[1];
  const float* b_qkv = (const float*)d_in[2];
  const float* w_o   = (const float*)d_in[3];
  const float* b_o   = (const float*)d_in[4];
  float* out = (float*)d_out;

  char* ws = (char*)d_ws;
  size_t off = 0;
  auto carve = [&](size_t bytes) -> void* {
    void* p = ws + off;
    off = (off + bytes + 255) & ~(size_t)255;
    return p;
  };
  __bf16* xb  = (__bf16*)carve((size_t)NBS * ND * 2);  // x in bf16
  __bf16* wqb = (__bf16*)carve((size_t)NE * ND * 2);   // w_qkv bf16
  __bf16* wob = (__bf16*)carve((size_t)ND * ND * 2);   // w_o bf16
  __bf16* qb  = (__bf16*)carve((size_t)NBS * ND * 2);  // Q [B,H,S,64] (pre-scaled)
  __bf16* kb  = (__bf16*)carve((size_t)NBS * ND * 2);  // K [B,H,S,64]
  __bf16* vb  = (__bf16*)carve((size_t)NBS * ND * 2);  // V [B,H,S,64]
  __bf16* atb = (__bf16*)carve((size_t)NBS * ND * 2);  // attn values [B,S,D]

  const int nx = NBS * ND, nwq = NE * ND, nwo = ND * ND;
  cvt_f32_bf16<<<(nx  + 255) / 256, 256, 0, stream>>>(x,     xb,  nx);
  cvt_f32_bf16<<<(nwq + 255) / 256, 256, 0, stream>>>(w_qkv, wqb, nwq);
  cvt_f32_bf16<<<(nwo + 255) / 256, 256, 0, stream>>>(w_o,   wob, nwo);

  qkv_gemm<<<dim3(NBS / 128, NE / 128), 256, 0, stream>>>(xb, wqb, b_qkv, qb, kb, vb);
  attn_kernel<<<dim3(NS / 128, NB * NH), 256, 0, stream>>>(qb, kb, vb, atb);
  out_gemm<<<dim3(NBS / 128, ND / 128), 256, 0, stream>>>(atb, wob, b_o, out);
}
